// GraphTransformer_77214922047499
// MI455X (gfx1250) — matrix operand order repaired and checked by reading, not compiled, not run on hardware
//
#include <hip/hip_runtime.h>
#include <hip/hip_bf16.h>

#define NNODES 50000
#define NEDGES 800000
#define CDIM   128

typedef __attribute__((ext_vector_type(16))) __bf16 v16bf;
typedef __attribute__((ext_vector_type(8)))  float  v8f;

union Frag { v16bf v; uint4 q[2]; };

__device__ inline unsigned short f2bf(float f) {
  unsigned int u = __float_as_uint(f);
  unsigned int r = (u + 0x7FFFu + ((u >> 16) & 1u)) >> 16;  // round-to-nearest-even
  return (unsigned short)r;
}

__device__ inline void atomicMaxF(float* addr, float v) {
  // monotonic int/uint encoding trick; init must be -INF (0xFF800000)
  if (v >= 0.f) atomicMax((int*)addr, __float_as_int(v));
  else          atomicMin((unsigned int*)addr, __float_as_uint(v));
}

// ---------------------------------------------------------------------------
// Weight prep: Wt[g*128+n][k] = Wg[k][n] (bf16, transposed so GEMM LDS reads
// are K-contiguous), biasc[g*128+n] = bg[n].
// ---------------------------------------------------------------------------
__global__ __launch_bounds__(256) void prep_weights(
    const float* __restrict__ Wq, const float* __restrict__ bq,
    const float* __restrict__ Wk, const float* __restrict__ bk,
    const float* __restrict__ Wv, const float* __restrict__ bv,
    const float* __restrict__ Ws, const float* __restrict__ bs,
    unsigned short* __restrict__ Wt, float* __restrict__ biasc)
{
  int o = blockIdx.x * 256 + threadIdx.x;   // 0 .. 65535
  int nall = o >> 7;                        // 0 .. 511
  int kk   = o & 127;
  int g    = nall >> 7;
  int n    = nall & 127;
  const float* W = (g == 0) ? Wq : (g == 1) ? Wk : (g == 2) ? Wv : Ws;
  Wt[(size_t)nall * CDIM + kk] = f2bf(W[kk * CDIM + n]);
  if (kk == 0) {
    const float* b = (g == 0) ? bq : (g == 1) ? bk : (g == 2) ? bv : bs;
    biasc[nall] = b[n];
  }
}

// ---------------------------------------------------------------------------
// Fused QKVS GEMM: [M,128] x [128,512] -> four [M,128] buffers.
// blockDim 256 (8 waves); block tile 64x128; wave tile 16x64.
// v_wmma_f32_16x16x32_bf16, K=128 in 4 chunks, 16 WMMA per wave.
// B^T tile staged with CDNA5 async global->LDS copies (ASYNCcnt).
// ---------------------------------------------------------------------------
__global__ __launch_bounds__(256) void qkvs_gemm(
    const float* __restrict__ A, const unsigned short* __restrict__ Wt,
    const float* __restrict__ biasc,
    float* __restrict__ dq, float* __restrict__ dk,
    float* __restrict__ dv, float* __restrict__ dout,
    int M, int relu)
{
  __shared__ unsigned short As[64 * 136];    // A tile, bf16, padded stride
  __shared__ unsigned short Bs[128 * 136];   // B^T tile (col-major weights)

  const int tid = threadIdx.x;
  const int mb  = blockIdx.x * 64;
  const int cb  = blockIdx.y;                // 0=Q 1=K 2=V 3=skip->out

  // --- stage B^T tile: async global->LDS b128 copies, no VGPR round-trip ---
  #pragma unroll
  for (int j = 0; j < 8; ++j) {
    int lin = j * 256 + tid;                 // 0..2047 uint4 slots
    int n   = lin >> 4;                      // 0..127
    int k16 = lin & 15;
    const unsigned short* gp = Wt + ((size_t)(cb * 128 + n)) * CDIM + k16 * 8;
    // low 32 bits of generic shared pointer == byte offset in LDS allocation
    unsigned int lds_addr = (unsigned int)(unsigned long long)&Bs[n * 136 + k16 * 8];
    asm volatile("global_load_async_to_lds_b128 %0, %1, off"
                 :: "v"(lds_addr), "v"((unsigned long long)gp)
                 : "memory");
  }

  // --- stage A tile: 64 rows x 128 cols fp32 -> bf16 (+optional ReLU) ---
  #pragma unroll
  for (int j = 0; j < 8; ++j) {
    int lin = j * 256 + tid;                 // 0..2047 float4 slots
    int r   = lin >> 5;                      // row 0..63
    int c4  = lin & 31;                      // float4 column
    int gi  = mb + r;
    float4 f = make_float4(0.f, 0.f, 0.f, 0.f);
    if (gi < M) f = *(const float4*)(A + (size_t)gi * CDIM + c4 * 4);
    if (relu) {
      f.x = fmaxf(f.x, 0.f); f.y = fmaxf(f.y, 0.f);
      f.z = fmaxf(f.z, 0.f); f.w = fmaxf(f.w, 0.f);
    }
    unsigned int p0 = (unsigned int)f2bf(f.x) | ((unsigned int)f2bf(f.y) << 16);
    unsigned int p1 = (unsigned int)f2bf(f.z) | ((unsigned int)f2bf(f.w) << 16);
    *(uint2*)&As[r * 136 + c4 * 4] = make_uint2(p0, p1);
  }

  // drain this wave's async copies, then workgroup barrier (orders ds stores)
  asm volatile("s_wait_asynccnt 0x0" ::: "memory");
  __syncthreads();

  const int wave    = tid >> 5;
  const int lane    = tid & 31;
  const int rowblk  = (wave & 3) * 16;       // 4 row blocks of 16
  const int colhalf = (wave >> 2) * 64;      // 2 col halves of 64
  const int lmod    = lane & 15;
  const int lhi     = lane >> 4;

  v8f acc[4];
  #pragma unroll
  for (int t = 0; t < 4; ++t)
    acc[t] = (v8f){0.f, 0.f, 0.f, 0.f, 0.f, 0.f, 0.f, 0.f};

  #pragma unroll
  for (int kc = 0; kc < 128; kc += 32) {
    // A fragment 16x32: lane m=lmod; lanes 0-15 K=kc+{0..7,16..23},
    // lanes 16-31 K=kc+{8..15,24..31}  (ISA 7.12.2)
    Frag a;
    int am = rowblk + lmod;
    int ak = kc + lhi * 8;
    a.q[0] = *(const uint4*)&As[am * 136 + ak];
    a.q[1] = *(const uint4*)&As[am * 136 + ak + 16];
    #pragma unroll
    for (int t = 0; t < 4; ++t) {
      // B fragment 32x16: lane col n=lmod; lanes 0-15 K=kc+{0..7,16..23},
      // lanes 16-31 K=kc+{8..15,24..31}, 2 K per VGPR
      Frag b;
      int bn = colhalf + t * 16 + lmod;
      int bk = kc + lhi * 8;
      b.q[0] = *(const uint4*)&Bs[bn * 136 + bk];
      b.q[1] = *(const uint4*)&Bs[bn * 136 + bk + 16];
      acc[t] = __builtin_amdgcn_wmma_f32_16x16x32_bf16(
          false, a.v, false, b.v, (short)0, acc[t], false, false);
    }
  }

  float* dst = (cb == 0) ? dq : (cb == 1) ? dk : (cb == 2) ? dv : dout;
  #pragma unroll
  for (int t = 0; t < 4; ++t) {
    int nl = colhalf + t * 16 + lmod;
    float bias = biasc[cb * 128 + nl];
    int mrow0 = mb + rowblk + lhi * 8;       // C/D: VGPR r -> M = r (+8 hi half)
    #pragma unroll
    for (int r = 0; r < 8; ++r) {
      int gm = mrow0 + r;
      if (gm < M) dst[(size_t)gm * CDIM + nl] = acc[t][r] + bias;
    }
  }
}

// ---------------------------------------------------------------------------
// Edge pass 1: alpha[e,h] = (q[dst]·k[src])/sqrt(D), segment max via atomics.
// ---------------------------------------------------------------------------
template <int H, int D>
__global__ __launch_bounds__(256) void edge_alpha(
    const int* __restrict__ src, const int* __restrict__ dst,
    const float* __restrict__ q, const float* __restrict__ k,
    float* __restrict__ alpha, float* __restrict__ amax)
{
  int idx = blockIdx.x * 256 + threadIdx.x;
  if (idx >= NEDGES * H) return;
  int e = idx / H, h = idx % H;
  int s = src[e], d = dst[e];
  const float4* qp = (const float4*)(q + (size_t)d * CDIM + h * D);
  const float4* kp = (const float4*)(k + (size_t)s * CDIM + h * D);
  float dot = 0.f;
  #pragma unroll
  for (int i = 0; i < D / 4; ++i) {
    float4 a = qp[i], b = kp[i];
    dot += a.x * b.x + a.y * b.y + a.z * b.z + a.w * b.w;
  }
  float val = dot * (1.0f / sqrtf((float)D));
  alpha[idx] = val;
  atomicMaxF(&amax[d * H + h], val);
}

// ---------------------------------------------------------------------------
// Edge pass 2: ex = exp(alpha - amax[dst]); segment sum via atomics.
// ---------------------------------------------------------------------------
template <int H>
__global__ __launch_bounds__(256) void edge_exp(
    const int* __restrict__ dst, float* __restrict__ alpha,
    const float* __restrict__ amax, float* __restrict__ denom)
{
  int idx = blockIdx.x * 256 + threadIdx.x;
  if (idx >= NEDGES * H) return;
  int e = idx / H, h = idx % H;
  int d = dst[e];
  float ex = __expf(alpha[idx] - amax[d * H + h]);
  alpha[idx] = ex;
  atomicAdd(&denom[d * H + h], ex);
}

// ---------------------------------------------------------------------------
// Edge pass 3: out[dst] += v[src] * (ex / denom[dst]).  One thread per
// (edge, float4-of-128-cols): H*D == 128 for every layer.
// ---------------------------------------------------------------------------
template <int H, int D>
__global__ __launch_bounds__(256) void edge_scatter(
    const int* __restrict__ src, const int* __restrict__ dst,
    const float* __restrict__ v, const float* __restrict__ alpha,
    const float* __restrict__ denom, float* __restrict__ out)
{
  int idx = blockIdx.x * 256 + threadIdx.x;
  if (idx >= NEDGES * 32) return;
  int e = idx >> 5;
  int c = idx & 31;                 // float4 column within 128
  int h = (c * 4) / D;
  int s = src[e], d = dst[e];
  float a = alpha[e * H + h] / fmaxf(denom[d * H + h], 1e-16f);
  float4 vv = *(const float4*)(v + (size_t)s * CDIM + c * 4);
  float* op = out + (size_t)d * CDIM + c * 4;
  atomicAdd(op + 0, vv.x * a);
  atomicAdd(op + 1, vv.y * a);
  atomicAdd(op + 2, vv.z * a);
  atomicAdd(op + 3, vv.w * a);
}

__global__ __launch_bounds__(256) void init_softmax(
    float* __restrict__ amax, float* __restrict__ denom, int n)
{
  int i = blockIdx.x * 256 + threadIdx.x;
  if (i < n) { amax[i] = __uint_as_float(0xFF800000u); denom[i] = 0.f; }
}

// ---------------------------------------------------------------------------
extern "C" void kernel_launch(void* const* d_in, const int* in_sizes, int n_in,
                              void* d_out, int out_size, void* d_ws, size_t ws_size,
                              hipStream_t stream) {
  (void)in_sizes; (void)n_in; (void)out_size; (void)ws_size;

  const float* x   = (const float*)d_in[0];
  const int*   ei  = (const int*)d_in[1];
  const int*   src = ei;
  const int*   dst = ei + NEDGES;

  // workspace carve-up (256B aligned)
  size_t off = 0;
  auto alloc = [&](size_t bytes) -> void* {
    void* p = (char*)d_ws + off;
    off = (off + bytes + 255) & ~(size_t)255;
    return p;
  };
  unsigned short* Wt    = (unsigned short*)alloc(512 * CDIM * sizeof(unsigned short));
  float* biasc = (float*)alloc(512 * sizeof(float));
  float* q     = (float*)alloc((size_t)NNODES * CDIM * sizeof(float));
  float* k     = (float*)alloc((size_t)NNODES * CDIM * sizeof(float));
  float* v     = (float*)alloc((size_t)NNODES * CDIM * sizeof(float));
  float* alpha = (float*)alloc((size_t)NEDGES * 4 * sizeof(float));
  float* amax  = (float*)alloc((size_t)NNODES * 4 * sizeof(float));
  float* denom = (float*)alloc((size_t)NNODES * 4 * sizeof(float));
  float* hA    = (float*)alloc((size_t)NNODES * CDIM * sizeof(float));
  float* hB    = (float*)alloc((size_t)NNODES * CDIM * sizeof(float));

  // per-layer param base in d_in: x, edge_index, then 10 entries per layer
  // (Wq,bq,Wk,bk,Wv,bv,Ws,bs,heads,d_head)
  auto layer = [&](int l, const float* Ain, int relu, float* outbuf, int H) {
    int b = 2 + 10 * l;
    prep_weights<<<256, 256, 0, stream>>>(
        (const float*)d_in[b + 0], (const float*)d_in[b + 1],
        (const float*)d_in[b + 2], (const float*)d_in[b + 3],
        (const float*)d_in[b + 4], (const float*)d_in[b + 5],
        (const float*)d_in[b + 6], (const float*)d_in[b + 7],
        Wt, biasc);
    dim3 g((NNODES + 63) / 64, 4);
    qkvs_gemm<<<g, 256, 0, stream>>>(Ain, Wt, biasc, q, k, v, outbuf,
                                     NNODES, relu);
    init_softmax<<<(NNODES * H + 255) / 256, 256, 0, stream>>>(amax, denom,
                                                               NNODES * H);
    int eh = NEDGES * H;
    if (H == 4) {
      edge_alpha<4, 32><<<(eh + 255) / 256, 256, 0, stream>>>(src, dst, q, k,
                                                              alpha, amax);
      edge_exp<4><<<(eh + 255) / 256, 256, 0, stream>>>(dst, alpha, amax, denom);
      edge_scatter<4, 32><<<(NEDGES * 32 + 255) / 256, 256, 0, stream>>>(
          src, dst, v, alpha, denom, outbuf);
    } else {
      edge_alpha<1, 128><<<(eh + 255) / 256, 256, 0, stream>>>(src, dst, q, k,
                                                               alpha, amax);
      edge_exp<1><<<(eh + 255) / 256, 256, 0, stream>>>(dst, alpha, amax, denom);
      edge_scatter<1, 128><<<(NEDGES * 32 + 255) / 256, 256, 0, stream>>>(
          src, dst, v, alpha, denom, outbuf);
    }
  };

  layer(0, x,  /*relu=*/0, hA, 4);            // relu applied at next A load
  layer(1, hA, /*relu=*/1, hB, 4);
  layer(2, hB, /*relu=*/1, (float*)d_out, 1); // final layer: no trailing relu
}
